// DAWNBlock_28930899706063
// MI455X (gfx1250) — compile-verified, matrix operand order checked by reading, not run on hardware
//
#include <hip/hip_runtime.h>

#define BB 4
#define SS 1024
#define DD 1024
#define HH 16
#define DH 64
#define RR 128
#define NCMP 16
#define NKN 32768
#define KTOP 8
#define BSS (BB*SS)

#if defined(__has_builtin)
#if __has_builtin(__builtin_amdgcn_global_load_async_to_lds_b128) && \
    __has_builtin(__builtin_amdgcn_s_wait_asynccnt)
#define HAVE_ASYNC_LDS 1
#endif
#if __has_builtin(__builtin_amdgcn_exp2f)
#define FAST_EXP2(x) __builtin_amdgcn_exp2f(x)
#endif
#if __has_builtin(__builtin_amdgcn_rcpf)
#define FAST_RCP(x) __builtin_amdgcn_rcpf(x)
#endif
#endif
#ifndef FAST_EXP2
#define FAST_EXP2(x) exp2f(x)
#endif
#ifndef FAST_RCP
#define FAST_RCP(x) (1.0f / (x))
#endif

typedef __attribute__((ext_vector_type(16))) __bf16 v16bf;
typedef __attribute__((ext_vector_type(8)))  float  v8f;
typedef int v4i __attribute__((vector_size(16)));
typedef __attribute__((address_space(1))) v4i* gptr_v4i;
typedef __attribute__((address_space(3))) v4i* lptr_v4i;

union Frag { v16bf v; unsigned u[8]; };

// hardware RNE f32->bf16 (v_cvt, 1 VALU op instead of a 3-op software round)
__device__ __forceinline__ unsigned short bf16u(float f) {
  return __builtin_bit_cast(unsigned short, (__bf16)f);
}
__device__ __forceinline__ unsigned pack2(float lo, float hi) {
  return (unsigned)bf16u(lo) | ((unsigned)bf16u(hi) << 16);
}
// CDNA5 16-bit A/B fragment K-offset pattern for VGPR v (0..7), lane-half (0/1):
// v<4 -> K = 2v (+8*half); v>=4 -> K = 16 + 2(v-4) (+8*half); pairs contiguous in K.
__device__ __forceinline__ int frag_k(int v, int half) {
  int base = (v < 4) ? (2 * v) : (16 + 2 * (v - 4));
  return base + 8 * half;
}
#define WMMA_BF16(a,b,c) \
  __builtin_amdgcn_wmma_f32_16x16x32_bf16(false,(a),false,(b),(short)0,(c),false,false)

// ---------------------------------------------------------------------------
// LayerNorm: one block per token row of 1024; outputs f32 + bf16 copies.
// ---------------------------------------------------------------------------
__global__ void __launch_bounds__(256) ln_kernel(const float* xin, const float* g,
                                                 const float* bta, float* hf,
                                                 unsigned short* hb) {
  __shared__ float r1[256], r2[256];
  int row = blockIdx.x, tid = threadIdx.x;
  const float* xr = xin + (long)row * DD;
  float s = 0.f, ss = 0.f;
  for (int d = tid; d < DD; d += 256) { float v = xr[d]; s += v; ss += v * v; }
  r1[tid] = s; r2[tid] = ss; __syncthreads();
  for (int o2 = 128; o2 > 0; o2 >>= 1) {
    if (tid < o2) { r1[tid] += r1[tid + o2]; r2[tid] += r2[tid + o2]; }
    __syncthreads();
  }
  float mu = r1[0] * (1.0f / DD);
  float var = r2[0] * (1.0f / DD) - mu * mu;
  float rs = rsqrtf(var + 1e-5f);
  for (int d = tid; d < DD; d += 256) {
    float v = (xr[d] - mu) * rs * g[d] + bta[d];
    hf[(long)row * DD + d] = v;
    hb[(long)row * DD + d] = bf16u(v);
  }
}

// ---------------------------------------------------------------------------
// Pack f32 [K,N] row-major -> bf16 pair-packed dwords: dst[(k/2)*N + n] = {k, k+1}
// ---------------------------------------------------------------------------
__global__ void __launch_bounds__(256) pack_kernel(const float* src, unsigned* dst,
                                                   int K, int N) {
  long pairs = (long)(K >> 1) * N;
  long i = (long)blockIdx.x * blockDim.x + threadIdx.x;
  if (i >= pairs) return;
  const float* s = src + (long)blockIdx.z * K * N;
  unsigned* d = dst + (long)blockIdx.z * pairs;
  long pr = i / N, n = i % N;
  d[pr * N + n] = pack2(s[(2 * pr) * N + n], s[(2 * pr + 1) * N + n]);
}

__global__ void __launch_bounds__(256) cvt_kernel(const float* src, unsigned short* dst,
                                                  long n) {
  long i = (long)blockIdx.x * blockDim.x + threadIdx.x;
  if (i < n) dst[i] = bf16u(src[i]);
}

// ---------------------------------------------------------------------------
// Router softmax: one wave per token.
// ---------------------------------------------------------------------------
__global__ void __launch_bounds__(32) router_kernel(const float* hf, const float* W,
                                                    float* wout) {
  int row = blockIdx.x, lane = threadIdx.x, col = lane & 15;
  const float* hr = hf + (long)row * DD;
  float s = 0.f;
  for (int d = 0; d < DD; ++d) s += hr[d] * W[d * NCMP + col];
  float mx = s;
  mx = fmaxf(mx, __shfl_xor(mx, 1, 32));
  mx = fmaxf(mx, __shfl_xor(mx, 2, 32));
  mx = fmaxf(mx, __shfl_xor(mx, 4, 32));
  mx = fmaxf(mx, __shfl_xor(mx, 8, 32));
  float e = __expf(s - mx);
  float sum = e;
  sum += __shfl_xor(sum, 1, 32);
  sum += __shfl_xor(sum, 2, 32);
  sum += __shfl_xor(sum, 4, 32);
  sum += __shfl_xor(sum, 8, 32);
  if (lane < 16) wout[(long)row * NCMP + lane] = e / sum;
}

// ---------------------------------------------------------------------------
// bf16 WMMA GEMM: C[M,N] = alpha * A[M,K] @ B[K,N] (+epilogue).
// EPI: 0=f32, 1=f32+residual, 2=bf16, 3=bf16 transposed to [b,h,dh,s] (for V).
// ---------------------------------------------------------------------------
template <int EPI>
__global__ void __launch_bounds__(128) gemm_kernel(const unsigned short* A,
                                                   const unsigned* Bp, float* Cf,
                                                   unsigned short* Cb,
                                                   const float* resid, int M, int N,
                                                   int K, int lda, long bStrB,
                                                   long bStrC, float alpha) {
  const int wave = threadIdx.x >> 5, lane = threadIdx.x & 31;
  const int half = lane >> 4, ln16 = lane & 15;
  const int mblk = blockIdx.y * 128 + (wave >> 1) * 64;
  const int nblk = blockIdx.x * 128 + (wave & 1) * 64;
  const unsigned* Au = (const unsigned*)A;
  const unsigned* Bpu = Bp + (long)blockIdx.z * bStrB;
  const int ldaH = lda >> 1;
  v8f acc[4][4] = {};
  for (int kb = 0; kb < K; kb += 32) {
    if (kb + 32 < K) {  // gfx1250 global_prefetch_b8 of next K-slab
      __builtin_prefetch(&Au[(long)(mblk + lane) * ldaH + ((kb + 32) >> 1)], 0, 0);
      __builtin_prefetch(&Au[(long)(mblk + 32 + lane) * ldaH + ((kb + 32) >> 1)], 0, 0);
      __builtin_prefetch(&Bpu[(long)((kb + 32) >> 1) * N + nblk + lane], 0, 0);
      __builtin_prefetch(&Bpu[(long)((kb + 48) >> 1) * N + nblk + lane], 0, 0);
    }
    Frag af[4], bfr[4];
#pragma unroll
    for (int mt = 0; mt < 4; ++mt)
#pragma unroll
      for (int v = 0; v < 8; ++v)
        af[mt].u[v] = Au[(long)(mblk + mt * 16 + ln16) * ldaH +
                         ((kb + frag_k(v, half)) >> 1)];
#pragma unroll
    for (int nt = 0; nt < 4; ++nt)
#pragma unroll
      for (int v = 0; v < 8; ++v)
        bfr[nt].u[v] = Bpu[(long)((kb + frag_k(v, half)) >> 1) * N +
                           (nblk + nt * 16 + ln16)];
#pragma unroll
    for (int mt = 0; mt < 4; ++mt)
#pragma unroll
      for (int nt = 0; nt < 4; ++nt)
        acc[mt][nt] = WMMA_BF16(af[mt].v, bfr[nt].v, acc[mt][nt]);
  }
#pragma unroll
  for (int mt = 0; mt < 4; ++mt)
#pragma unroll
    for (int nt = 0; nt < 4; ++nt)
#pragma unroll
      for (int v = 0; v < 8; ++v) {
        int row = mblk + mt * 16 + half * 8 + v;
        int col = nblk + nt * 16 + ln16;
        long idx = (long)blockIdx.z * bStrC + (long)row * N + col;
        float val = acc[mt][nt][v] * alpha;
        if (EPI == 0)      Cf[idx] = val;
        else if (EPI == 1) Cf[idx] = resid[idx] + val;
        else if (EPI == 2) Cb[idx] = bf16u(val);
        else {  // EPI==3: V transposed to [b, h, dh, s]
          int bb2 = row >> 10, s2 = row & (SS - 1);
          int hh2 = col >> 6,  dh2 = col & (DH - 1);
          long tidx = ((long)((bb2 * HH + hh2) * DH + dh2)) * SS + s2;
          Cb[tidx] = bf16u(val);
        }
      }
}

// ---------------------------------------------------------------------------
// Weighted sums over the 16 shared compress projections.
// ---------------------------------------------------------------------------
__global__ void __launch_bounds__(128) wsum3_kernel(const float* proj, const float* wQ,
                                                    const float* wK, const float* wV,
                                                    unsigned short* hQ,
                                                    unsigned short* hK,
                                                    unsigned short* hV) {
  __shared__ float w[3][NCMP];
  int row = blockIdx.x, r = threadIdx.x;
  if (threadIdx.x < 16)       w[0][threadIdx.x]      = wQ[(long)row * NCMP + threadIdx.x];
  else if (threadIdx.x < 32)  w[1][threadIdx.x - 16] = wK[(long)row * NCMP + threadIdx.x - 16];
  else if (threadIdx.x < 48)  w[2][threadIdx.x - 32] = wV[(long)row * NCMP + threadIdx.x - 32];
  __syncthreads();
  float aq = 0.f, ak = 0.f, av = 0.f;
#pragma unroll
  for (int n = 0; n < NCMP; ++n) {
    float p = proj[((long)n * BSS + row) * RR + r];
    aq += w[0][n] * p; ak += w[1][n] * p; av += w[2][n] * p;
  }
  hQ[(long)row * RR + r] = bf16u(aq);
  hK[(long)row * RR + r] = bf16u(ak);
  hV[(long)row * RR + r] = bf16u(av);
}

// wsum1 folds the knowledge-score scale 1/sqrt(RANK) into Qm.
__global__ void __launch_bounds__(128) wsum1_kernel(const float* proj, const float* wM,
                                                    unsigned short* hM) {
  __shared__ float w[NCMP];
  int row = blockIdx.x, r = threadIdx.x;
  if (threadIdx.x < 16) w[threadIdx.x] = wM[(long)row * NCMP + threadIdx.x];
  __syncthreads();
  float a = 0.f;
#pragma unroll
  for (int n = 0; n < NCMP; ++n) a += w[n] * proj[((long)n * BSS + row) * RR + r];
  hM[(long)row * RR + r] = bf16u(a * 0.08838834764831845f);
}

// ---------------------------------------------------------------------------
// Flash attention, causal, bf16 WMMA. Scores arrive pre-scaled by
// (1/sqrt(dh))*log2(e), softmax runs in base-2 via raw v_exp_f32. Row sums of
// P come from a WMMA with a ones-fragment (C layout replicates each row's sum
// across the 16 lanes of its half).
// ---------------------------------------------------------------------------
__global__ void __launch_bounds__(128) attn_kernel(const unsigned short* Qb,
                                                   const unsigned short* Kb,
                                                   const unsigned short* Vt,
                                                   unsigned short* Ob) {
  __shared__ unsigned short Plds[4][64][68];
  const int wave = threadIdx.x >> 5, lane = threadIdx.x & 31;
  const int half = lane >> 4, ln16 = lane & 15;
  const int tiles = SS / 64;
  const int b = blockIdx.x / tiles, qt = blockIdx.x % tiles;
  const int h = blockIdx.y * 4 + wave;
  const unsigned* Qu = (const unsigned*)Qb;
  const unsigned* Ku = (const unsigned*)Kb;
  const unsigned* Vtu = (const unsigned*)Vt;
  const long vhead = ((long)(b * HH + h) * DH) * SS;
  const int qbase = qt * 64;

  Frag onesf;
#pragma unroll
  for (int v = 0; v < 8; ++v) onesf.u[v] = 0x3F803F80u;  // bf16 1.0 pair

  Frag qf[4][2];
#pragma unroll
  for (int mt = 0; mt < 4; ++mt)
#pragma unroll
    for (int ks = 0; ks < 2; ++ks)
#pragma unroll
      for (int v = 0; v < 8; ++v) {
        long e = ((long)(b * SS + qbase + mt * 16 + ln16)) * DD + h * DH +
                 ks * 32 + frag_k(v, half);
        qf[mt][ks].u[v] = Qu[e >> 1];
      }

  v8f o[4][4] = {};
  float mrow[4][8], lrow[4][8];
#pragma unroll
  for (int mt = 0; mt < 4; ++mt)
#pragma unroll
    for (int v = 0; v < 8; ++v) { mrow[mt][v] = -3.0e38f; lrow[mt][v] = 0.f; }

  for (int kt = 0; kt <= qt; ++kt) {
    const int kbase = kt * 64;
    v8f sa[4][4] = {};
#pragma unroll
    for (int ks = 0; ks < 2; ++ks) {
      Frag kf[4];
#pragma unroll
      for (int nt = 0; nt < 4; ++nt)
#pragma unroll
        for (int v = 0; v < 8; ++v) {
          long e = ((long)(b * SS + kbase + nt * 16 + ln16)) * DD + h * DH +
                   ks * 32 + frag_k(v, half);
          kf[nt].u[v] = Ku[e >> 1];
        }
#pragma unroll
      for (int mt = 0; mt < 4; ++mt)
#pragma unroll
        for (int nt = 0; nt < 4; ++nt)
          sa[mt][nt] = WMMA_BF16(qf[mt][ks].v, kf[nt].v, sa[mt][nt]);
    }
    if (kt == qt) {  // causal mask on the diagonal tile
#pragma unroll
      for (int mt = 0; mt < 4; ++mt)
#pragma unroll
        for (int nt = 0; nt < 4; ++nt)
#pragma unroll
          for (int v = 0; v < 8; ++v) {
            int rowi = mt * 16 + half * 8 + v;
            int coli = nt * 16 + ln16;
            if (coli > rowi) sa[mt][nt][v] = -3.0e38f;
          }
    }
    // online softmax (base-2): max via shfl, exp via raw v_exp_f32
#pragma unroll
    for (int mt = 0; mt < 4; ++mt)
#pragma unroll
      for (int v = 0; v < 8; ++v) {
        float rm = -3.0e38f;
#pragma unroll
        for (int nt = 0; nt < 4; ++nt) rm = fmaxf(rm, sa[mt][nt][v]);
        rm = fmaxf(rm, __shfl_xor(rm, 1, 32));
        rm = fmaxf(rm, __shfl_xor(rm, 2, 32));
        rm = fmaxf(rm, __shfl_xor(rm, 4, 32));
        rm = fmaxf(rm, __shfl_xor(rm, 8, 32));
        float mold = mrow[mt][v];
        float mnew = fmaxf(mold, rm);
        float corr = FAST_EXP2(mold - mnew);
        mrow[mt][v] = mnew;
#pragma unroll
        for (int nt = 0; nt < 4; ++nt)
          sa[mt][nt][v] = FAST_EXP2(sa[mt][nt][v] - mnew);
        lrow[mt][v] *= corr;
#pragma unroll
        for (int nt = 0; nt < 4; ++nt) o[mt][nt][v] *= corr;
      }
    // P (C layout) -> LDS bf16 -> A-layout frags
#pragma unroll
    for (int mt = 0; mt < 4; ++mt)
#pragma unroll
      for (int nt = 0; nt < 4; ++nt)
#pragma unroll
        for (int v = 0; v < 8; ++v)
          Plds[wave][mt * 16 + half * 8 + v][nt * 16 + ln16] = bf16u(sa[mt][nt][v]);
    __syncthreads();
    const unsigned* Pu = (const unsigned*)&Plds[wave][0][0];
    v8f rs[4] = {};
#pragma unroll
    for (int ks2 = 0; ks2 < 2; ++ks2) {
      Frag pf[4], vf[4];
#pragma unroll
      for (int mt = 0; mt < 4; ++mt)
#pragma unroll
        for (int v = 0; v < 8; ++v)
          pf[mt].u[v] = Pu[(mt * 16 + ln16) * 34 + ((ks2 * 32 + frag_k(v, half)) >> 1)];
#pragma unroll
      for (int nt2 = 0; nt2 < 4; ++nt2)
#pragma unroll
        for (int v = 0; v < 8; ++v) {
          int dh = nt2 * 16 + ln16;
          int k0 = ks2 * 32 + frag_k(v, half);
          vf[nt2].u[v] = Vtu[(vhead + (long)dh * SS + kbase + k0) >> 1];
        }
#pragma unroll
      for (int mt = 0; mt < 4; ++mt)
        rs[mt] = WMMA_BF16(pf[mt].v, onesf.v, rs[mt]);  // row sums of P
#pragma unroll
      for (int mt = 0; mt < 4; ++mt)
#pragma unroll
        for (int nt2 = 0; nt2 < 4; ++nt2)
          o[mt][nt2] = WMMA_BF16(pf[mt].v, vf[nt2].v, o[mt][nt2]);
    }
#pragma unroll
    for (int mt = 0; mt < 4; ++mt)
#pragma unroll
      for (int v = 0; v < 8; ++v) lrow[mt][v] += rs[mt][v];
    __syncthreads();
  }
  // normalize and store bf16
#pragma unroll
  for (int mt = 0; mt < 4; ++mt)
#pragma unroll
    for (int v = 0; v < 8; ++v) {
      float invl = FAST_RCP(lrow[mt][v]);
#pragma unroll
      for (int nt2 = 0; nt2 < 4; ++nt2) {
        long e = ((long)(b * SS + qbase + mt * 16 + half * 8 + v)) * DD + h * DH +
                 nt2 * 16 + ln16;
        Ob[e] = bf16u(o[mt][nt2][v] * invl);
      }
    }
}

// ---------------------------------------------------------------------------
// Fused knowledge scores (WMMA) + top-8 + softmax + V gather + residual.
// knowledge_K tiles are DOUBLE-BUFFERED through LDS with async copy-to-LDS:
// chunk n+1 is issued before waiting (s_wait_asynccnt 16 leaves it in
// flight), overlapping the L2 transfer with WMMA + scan of chunk n.
// ---------------------------------------------------------------------------
__global__ void __launch_bounds__(64) ktopk_kernel(const unsigned short* Qmb,
                                                   const unsigned short* kKb,
                                                   const float* kV, const float* x2,
                                                   float* out) {
  __shared__ unsigned Bt[2][64 * (RR >> 1)];  // 2 x 16 KB ping-pong
  __shared__ float Slds[2][16][65];
  __shared__ float wLds[32][KTOP];
  __shared__ int   iLds[32][KTOP];
  const int wave = threadIdx.x >> 5, lane = threadIdx.x & 31;
  const int half = lane >> 4, ln16 = lane & 15;
  const int rowbase = blockIdx.x * 32 + wave * 16;
  const unsigned* Qu = (const unsigned*)Qmb;
  const unsigned* Ku = (const unsigned*)kKb;

  auto stage = [&](int chunk, int buf) {  // copy rows chunk..chunk+63 -> Bt[buf]
#if HAVE_ASYNC_LDS
    const char* gsrc = (const char*)kKb + (long)(chunk + threadIdx.x) * (RR * 2);
    char* ldst = (char*)&Bt[buf][0] + threadIdx.x * (RR * 2);
#pragma unroll
    for (int i = 0; i < 16; ++i)
      __builtin_amdgcn_global_load_async_to_lds_b128(
          (gptr_v4i)(gsrc + i * 16), (lptr_v4i)(ldst + i * 16), 0, 0);
#else
    for (int i = threadIdx.x; i < 64 * (RR >> 1); i += 64)
      Bt[buf][i] = Ku[(long)(chunk + (i >> 6)) * (RR >> 1) + (i & 63)];
#endif
  };

  Frag af[4];
#pragma unroll
  for (int ks = 0; ks < 4; ++ks)
#pragma unroll
    for (int v = 0; v < 8; ++v)
      af[ks].u[v] = Qu[((long)(rowbase + ln16)) * (RR >> 1) +
                       ((ks * 32 + frag_k(v, half)) >> 1)];

  float val8[KTOP]; int idx8[KTOP];
#pragma unroll
  for (int k = 0; k < KTOP; ++k) { val8[k] = -3.0e38f; idx8[k] = 0; }

  stage(0, 0);  // prologue
  for (int nk = 0; nk < NKN; nk += 64) {
    const int cur = (nk >> 6) & 1;
    const bool hasNext = (nk + 64 < NKN);
    if (hasNext) stage(nk + 64, cur ^ 1);
#if HAVE_ASYNC_LDS
    if (hasNext) __builtin_amdgcn_s_wait_asynccnt(16);  // current chunk done
    else         __builtin_amdgcn_s_wait_asynccnt(0);
#endif
    if (nk + 128 < NKN)  // gfx1250 global_prefetch_b8, two chunks ahead
      __builtin_prefetch(&Ku[(long)(nk + 128 + threadIdx.x) * (RR >> 1)], 0, 0);
    __syncthreads();

    v8f acc[4] = {};
#pragma unroll
    for (int nt = 0; nt < 4; ++nt)
#pragma unroll
      for (int ks = 0; ks < 4; ++ks) {
        Frag bfr;
#pragma unroll
        for (int v = 0; v < 8; ++v)
          bfr.u[v] = Bt[cur][(nt * 16 + ln16) * (RR >> 1) +
                            ((ks * 32 + frag_k(v, half)) >> 1)];
        acc[nt] = WMMA_BF16(af[ks].v, bfr.v, acc[nt]);
      }
#pragma unroll
    for (int nt = 0; nt < 4; ++nt)
#pragma unroll
      for (int v = 0; v < 8; ++v)
        Slds[wave][half * 8 + v][nt * 16 + ln16] = acc[nt][v];
    __syncthreads();
    {  // all 32 lanes scan: 2 lanes per row, 32 columns each
      const int cbase = half * 32;
      for (int c = 0; c < 32; ++c) {
        float s = Slds[wave][ln16][cbase + c];
        if (s > val8[KTOP - 1]) {
          val8[KTOP - 1] = s; idx8[KTOP - 1] = nk + cbase + c;
#pragma unroll
          for (int j = KTOP - 1; j > 0; --j)
            if (val8[j] > val8[j - 1]) {
              float tv = val8[j]; val8[j] = val8[j - 1]; val8[j - 1] = tv;
              int ti = idx8[j]; idx8[j] = idx8[j - 1]; idx8[j - 1] = ti;
            }
        }
      }
    }
    __syncthreads();
  }
  {  // merge partner half's top-8 (snapshot first, then insert)
    float pv[KTOP]; int pi[KTOP];
#pragma unroll
    for (int k = 0; k < KTOP; ++k) {
      pv[k] = __shfl_xor(val8[k], 16, 32);
      pi[k] = __shfl_xor(idx8[k], 16, 32);
    }
#pragma unroll
    for (int k = 0; k < KTOP; ++k) {
      if (pv[k] > val8[KTOP - 1]) {
        val8[KTOP - 1] = pv[k]; idx8[KTOP - 1] = pi[k];
#pragma unroll
        for (int j = KTOP - 1; j > 0; --j)
          if (val8[j] > val8[j - 1]) {
            float tv = val8[j]; val8[j] = val8[j - 1]; val8[j - 1] = tv;
            int ti = idx8[j]; idx8[j] = idx8[j - 1]; idx8[j - 1] = ti;
          }
      }
    }
  }
  if (lane < 16) {
    float mx = val8[0], sum = 0.f, w[KTOP];
#pragma unroll
    for (int k = 0; k < KTOP; ++k) { w[k] = __expf(val8[k] - mx); sum += w[k]; }
    float inv = FAST_RCP(sum);
#pragma unroll
    for (int k = 0; k < KTOP; ++k) {
      wLds[wave * 16 + lane][k] = w[k] * inv;
      iLds[wave * 16 + lane][k] = idx8[k];
    }
  }
  __syncthreads();
  for (int r = 0; r < 32; ++r) {
    long row = (long)blockIdx.x * 32 + r;
    for (int d = threadIdx.x; d < DD; d += 64) {
      float a = x2[row * DD + d];
#pragma unroll
      for (int k = 0; k < KTOP; ++k)
        a += wLds[r][k] * kV[(long)iLds[r][k] * DD + d];
      out[row * DD + d] = a;
    }
  }
}

// ---------------------------------------------------------------------------
extern "C" void kernel_launch(void* const* d_in, const int* in_sizes, int n_in,
                              void* d_out, int out_size, void* d_ws, size_t ws_size,
                              hipStream_t stream) {
  const float* x       = (const float*)d_in[0];
  // d_in[1] = mask (causal, implicit in attn_kernel)
  const float* neurons = (const float*)d_in[2];
  const float* kK      = (const float*)d_in[3];
  const float* kV      = (const float*)d_in[4];
  const float* rQ      = (const float*)d_in[5];
  const float* rK      = (const float*)d_in[6];
  const float* rV      = (const float*)d_in[7];
  const float* rM      = (const float*)d_in[8];
  const float* WQ      = (const float*)d_in[9];
  const float* WK      = (const float*)d_in[10];
  const float* WV      = (const float*)d_in[11];
  const float* WO      = (const float*)d_in[12];
  const float* ln1g    = (const float*)d_in[13];
  const float* ln1b    = (const float*)d_in[14];
  const float* ln2g    = (const float*)d_in[15];
  const float* ln2b    = (const float*)d_in[16];
  float* out = (float*)d_out;

  unsigned char* base = (unsigned char*)d_ws;
  size_t off = 0;
  auto alloc = [&](size_t bytes) -> void* {
    void* p = base + off;
    off += (bytes + 255) & ~(size_t)255;
    return p;
  };
  float*          h_f  = (float*)alloc((size_t)BSS * DD * 4);
  unsigned short* h_b  = (unsigned short*)alloc((size_t)BSS * DD * 2);
  unsigned*       npk  = (unsigned*)alloc((size_t)NCMP * (DD / 2) * RR * 4);
  unsigned*       wqp  = (unsigned*)alloc((size_t)(RR / 2) * DD * 4);
  unsigned*       wkp  = (unsigned*)alloc((size_t)(RR / 2) * DD * 4);
  unsigned*       wvp  = (unsigned*)alloc((size_t)(RR / 2) * DD * 4);
  unsigned*       wop  = (unsigned*)alloc((size_t)(DD / 2) * DD * 4);
  unsigned short* kKb  = (unsigned short*)alloc((size_t)NKN * RR * 2);
  float*          wQw  = (float*)alloc((size_t)BSS * NCMP * 4);
  float*          wKw  = (float*)alloc((size_t)BSS * NCMP * 4);
  float*          wVw  = (float*)alloc((size_t)BSS * NCMP * 4);
  float*          wMw  = (float*)alloc((size_t)BSS * NCMP * 4);
  float*          proj = (float*)alloc((size_t)NCMP * BSS * RR * 4);
  unsigned short* hQb  = (unsigned short*)alloc((size_t)BSS * RR * 2);
  unsigned short* hKb  = (unsigned short*)alloc((size_t)BSS * RR * 2);
  unsigned short* hVb  = (unsigned short*)alloc((size_t)BSS * RR * 2);
  unsigned short* hMb  = (unsigned short*)alloc((size_t)BSS * RR * 2);
  unsigned short* Qbf  = (unsigned short*)alloc((size_t)BSS * DD * 2);
  unsigned short* Kbf  = (unsigned short*)alloc((size_t)BSS * DD * 2);
  unsigned short* Vtb  = (unsigned short*)alloc((size_t)BSS * DD * 2);
  unsigned short* attb = (unsigned short*)alloc((size_t)BSS * DD * 2);
  float*          x2   = (float*)alloc((size_t)BSS * DD * 4);
  float*          h2f  = (float*)alloc((size_t)BSS * DD * 4);
  unsigned short* h2b  = (unsigned short*)alloc((size_t)BSS * DD * 2);
  (void)in_sizes; (void)n_in; (void)out_size; (void)ws_size;

  // 1. LN1
  ln_kernel<<<BSS, 256, 0, stream>>>(x, ln1g, ln1b, h_f, h_b);
  // 2. weight conversion / packing
  pack_kernel<<<dim3(256, 1, NCMP), 256, 0, stream>>>(neurons, npk, DD, RR);
  pack_kernel<<<dim3(256, 1, 1), 256, 0, stream>>>(WQ, wqp, RR, DD);
  pack_kernel<<<dim3(256, 1, 1), 256, 0, stream>>>(WK, wkp, RR, DD);
  pack_kernel<<<dim3(256, 1, 1), 256, 0, stream>>>(WV, wvp, RR, DD);
  pack_kernel<<<dim3(2048, 1, 1), 256, 0, stream>>>(WO, wop, DD, DD);
  cvt_kernel<<<(int)(((long)NKN * RR + 255) / 256), 256, 0, stream>>>(kK, kKb,
                                                                      (long)NKN * RR);
  // 3. routers (Q/K/V share h)
  router_kernel<<<BSS, 32, 0, stream>>>(h_f, rQ, wQw);
  router_kernel<<<BSS, 32, 0, stream>>>(h_f, rK, wKw);
  router_kernel<<<BSS, 32, 0, stream>>>(h_f, rV, wVw);
  // 4. all_proj (computed ONCE, shared by Q/K/V routers)
  gemm_kernel<0><<<dim3(1, BSS / 128, NCMP), 128, 0, stream>>>(
      h_b, npk, proj, nullptr, nullptr, BSS, RR, DD, DD,
      (long)(DD / 2) * RR, (long)BSS * RR, 1.0f);
  wsum3_kernel<<<BSS, RR, 0, stream>>>(proj, wQw, wKw, wVw, hQb, hKb, hVb);
  // 5. Q/K/V projections. Q scaled by (1/sqrt(dh))*log2(e) for base-2 softmax.
  gemm_kernel<2><<<dim3(DD / 128, BSS / 128, 1), 128, 0, stream>>>(
      hQb, wqp, nullptr, Qbf, nullptr, BSS, DD, RR, RR, 0, 0, 0.18033688011112042f);
  gemm_kernel<2><<<dim3(DD / 128, BSS / 128, 1), 128, 0, stream>>>(
      hKb, wkp, nullptr, Kbf, nullptr, BSS, DD, RR, RR, 0, 0, 1.0f);
  gemm_kernel<3><<<dim3(DD / 128, BSS / 128, 1), 128, 0, stream>>>(
      hVb, wvp, nullptr, Vtb, nullptr, BSS, DD, RR, RR, 0, 0, 1.0f);
  // 6. causal flash attention
  attn_kernel<<<dim3(BB * (SS / 64), HH / 4), 128, 0, stream>>>(Qbf, Kbf, Vtb, attb);
  // 7. output projection + residual
  gemm_kernel<1><<<dim3(DD / 128, BSS / 128, 1), 128, 0, stream>>>(
      attb, wop, x2, nullptr, x, BSS, DD, DD, DD, 0, 0, 1.0f);
  // 8. LN2 + M-router + all_proj on h2 + weighted sum
  ln_kernel<<<BSS, 256, 0, stream>>>(x2, ln2g, ln2b, h2f, h2b);
  router_kernel<<<BSS, 32, 0, stream>>>(h2f, rM, wMw);
  gemm_kernel<0><<<dim3(1, BSS / 128, NCMP), 128, 0, stream>>>(
      h2b, npk, proj, nullptr, nullptr, BSS, RR, DD, DD,
      (long)(DD / 2) * RR, (long)BSS * RR, 1.0f);
  wsum1_kernel<<<BSS, RR, 0, stream>>>(proj, wMw, hMb);
  // 9. fused knowledge scores + top-8 + softmax + gather + residual
  ktopk_kernel<<<BSS / 32, 64, 0, stream>>>(hMb, kKb, kV, x2, out);
}
